// selfAttention_17386027614596
// MI455X (gfx1250) — matrix operand order repaired and checked by reading, not compiled, not run on hardware
//
#include <hip/hip_runtime.h>

// ---------------------------------------------------------------------------
// Self-attention block (GroupNorm -> QKV proj -> softmax attention -> out proj
// + residual) for B=2, C=256, H=W=64, 4 heads, head_dim 64, on gfx1250.
// All GEMM-like math runs on v_wmma_f32_16x16x32_f16 (wave32 WMMA).
// Flash-attention K/V tiles are staged in LDS with async global->LDS copies
// (double buffered, ASYNCcnt) when the toolchain exposes the builtins.
// ---------------------------------------------------------------------------

typedef __attribute__((ext_vector_type(16))) _Float16 v16h;
typedef __attribute__((ext_vector_type(8)))  _Float16 h8;
typedef __attribute__((ext_vector_type(8)))  float    v8f;
typedef __attribute__((__vector_size__(16))) int      vsi4;   // 16B chunk type

#define BATCH   2
#define CHN     256
#define LSP     4096      // 64*64 spatial positions
#define NH      4
#define HD      64
#define GROUPS  32
#define CPG     8         // channels per group

// Workspace layout (units: _Float16 elements)
#define OFF_WQ  0u
#define OFF_WK  65536u
#define OFF_WV  131072u
#define OFF_WO  196608u
#define OFF_XN  262144u                 // [B][L][C]      2*4096*256
#define OFF_Q   (OFF_XN + 2097152u)     // [B][NH][L][HD]
#define OFF_K   (OFF_Q  + 2097152u)     // [B][NH][L][HD]
#define OFF_V   (OFF_K  + 2097152u)     // [B][NH][HD][L]
#define OFF_AO  (OFF_V  + 2097152u)     // [B][L][C]

#define WMMA_F16(a, b, c) \
  __builtin_amdgcn_wmma_f32_16x16x32_f16(false, (a), false, (b), (short)0, (c), false, false)

// --- async global->LDS support (guarded: falls back to register staging) ---
#if defined(__has_builtin)
#if __has_builtin(__builtin_amdgcn_global_load_async_to_lds_b128) && \
    __has_builtin(__builtin_amdgcn_s_wait_asynccnt)
#define USE_ASYNC_LDS 1
#endif
#endif
#ifndef USE_ASYNC_LDS
#define USE_ASYNC_LDS 0
#endif

// Copy 16 bytes global -> LDS (per lane). Async variant uses ASYNCcnt.
__device__ __forceinline__ void copy16_g2l(const _Float16* g, _Float16* l) {
#if USE_ASYNC_LDS
  __builtin_amdgcn_global_load_async_to_lds_b128(
      (__attribute__((address_space(1))) vsi4*)(uintptr_t)g,
      (__attribute__((address_space(3))) vsi4*)(unsigned int)(uintptr_t)l,
      0, 0);
#else
  *(h8*)l = *(const h8*)g;
#endif
}

template <int N>
__device__ __forceinline__ void wait_async() {
#if USE_ASYNC_LDS
  __builtin_amdgcn_s_wait_asynccnt((unsigned short)N);
#endif
}

// A-fragment (16x32 f16, M x K): lane holds row = lane%16;
// halves are K = 16*hi .. 16*hi+15  (hi = lane>=16)
__device__ __forceinline__ v16h load_afrag(const _Float16* __restrict__ row_k0, int hi) {
  v16h a;
  *(h8*)&a       = *(const h8*)(row_k0 + 16 * hi);
  *((h8*)&a + 1) = *(const h8*)(row_k0 + 16 * hi + 8);
  return a;
}

// B-fragment (32x16 f16, K x N): lane holds col = lane%16;
// halves are K = 16*hi .. 16*hi+15 (caller folds the 16*hi into the pointer)
__device__ __forceinline__ v16h load_bfrag(const _Float16* __restrict__ col_k) {
  v16h b;
  *(h8*)&b       = *(const h8*)(col_k);
  *((h8*)&b + 1) = *(const h8*)(col_k + 8);
  return b;
}

// ---------------------------------------------------------------------------
// 1) Convert the four 256x256 f32 weight matrices to f16 in workspace
// ---------------------------------------------------------------------------
__global__ void cvt_weights_kernel(const float* __restrict__ wq,
                                   const float* __restrict__ wk,
                                   const float* __restrict__ wv,
                                   const float* __restrict__ wo,
                                   _Float16* __restrict__ out) {
  int i = blockIdx.x * 256 + threadIdx.x;          // 0 .. 4*65536-1
  const float* src = (i < 65536) ? wq : (i < 131072) ? wk : (i < 196608) ? wv : wo;
  out[i] = (_Float16)src[i & 65535];
}

// ---------------------------------------------------------------------------
// 2) GroupNorm -> normalized activations, f16, TRANSPOSED to [b][l][c]
//    (that is exactly the B-matrix ([N][K]) layout the WMMA GEMMs want)
// ---------------------------------------------------------------------------
__global__ void groupnorm_kernel(const float* __restrict__ x,
                                 const float* __restrict__ gw,
                                 const float* __restrict__ gb,
                                 _Float16* __restrict__ xn) {
  __shared__ float rs[256];
  __shared__ float rq[256];
  int blk = blockIdx.x;
  int b = blk >> 5, g = blk & 31;
  const float* xb = x + ((size_t)b * CHN + g * CPG) * LSP;
  const int n = CPG * LSP;                          // 32768 elements per group
  float s = 0.f, s2 = 0.f;
  for (int i = threadIdx.x; i < n; i += 256) {
    float v = xb[i];
    s += v; s2 += v * v;
  }
  rs[threadIdx.x] = s; rq[threadIdx.x] = s2;
  __syncthreads();
  for (int off = 128; off > 0; off >>= 1) {
    if ((int)threadIdx.x < off) {
      rs[threadIdx.x] += rs[threadIdx.x + off];
      rq[threadIdx.x] += rq[threadIdx.x + off];
    }
    __syncthreads();
  }
  float mean = rs[0] / (float)n;
  float var  = rq[0] / (float)n - mean * mean;
  float inv  = rsqrtf(var + 1e-5f);
  for (int i = threadIdx.x; i < n; i += 256) {
    int cl = i >> 12;                               // i / 4096
    int l  = i & 4095;
    int c  = g * CPG + cl;
    float v = (xb[i] - mean) * inv * gw[c] + gb[c];
    xn[((size_t)b * LSP + l) * CHN + c] = (_Float16)v;
  }
}

// ---------------------------------------------------------------------------
// 3) QKV projection: D[m][l] = W[m][k] * Xn[l][k]   (K = 256)
//    mode 0: store [b][head][l][hd]   (Q and K layout; A-fragment friendly)
//    mode 1: store [b][head][hd][L]   (V layout; B-fragment friendly for P*V)
//    Block = 4 waves; each wave owns a 32(M) x 32(N) tile; block = 64x64.
// ---------------------------------------------------------------------------
__global__ void qkv_gemm_kernel(const _Float16* __restrict__ W,
                                const _Float16* __restrict__ Xn,
                                _Float16* __restrict__ Out,
                                int mode) {
  int wave = threadIdx.x >> 5, lane = threadIdx.x & 31;
  int hi = lane >> 4, ln = lane & 15;
  int b  = blockIdx.z;
  int mb = blockIdx.y * 64 + (wave & 1) * 32;
  int nb = blockIdx.x * 64 + (wave >> 1) * 32;
  const _Float16* Xb = Xn + (size_t)b * LSP * CHN;

  v8f acc[2][2] = {};
  for (int k0 = 0; k0 < CHN; k0 += 32) {
    v16h a0 = load_afrag(W + (size_t)(mb + ln) * CHN + k0, hi);
    v16h a1 = load_afrag(W + (size_t)(mb + 16 + ln) * CHN + k0, hi);
    v16h b0 = load_bfrag(Xb + (size_t)(nb + ln) * CHN + k0 + 16 * hi);
    v16h b1 = load_bfrag(Xb + (size_t)(nb + 16 + ln) * CHN + k0 + 16 * hi);
    acc[0][0] = WMMA_F16(a0, b0, acc[0][0]);
    acc[0][1] = WMMA_F16(a0, b1, acc[0][1]);
    acc[1][0] = WMMA_F16(a1, b0, acc[1][0]);
    acc[1][1] = WMMA_F16(a1, b1, acc[1][1]);
  }

#pragma unroll
  for (int mi = 0; mi < 2; ++mi)
#pragma unroll
    for (int ni = 0; ni < 2; ++ni) {
      int m0 = mb + mi * 16 + 8 * hi;               // 8 consecutive rows per lane
      int n  = nb + ni * 16 + ln;
      if (mode == 0) {
        // rows m0..m0+7 -> contiguous halves at [b][head][n][cl..cl+7]
        int head = m0 >> 6, cl = m0 & 63;
        h8 v;
#pragma unroll
        for (int r = 0; r < 8; ++r) v[r] = (_Float16)acc[mi][ni][r];
        *(h8*)(Out + (((size_t)b * NH + head) * (size_t)LSP + n) * HD + cl) = v;
      } else {
#pragma unroll
        for (int r = 0; r < 8; ++r) {
          int m = m0 + r;
          Out[(((size_t)b * NH + (m >> 6)) * (size_t)HD + (m & 63)) * LSP + n] =
              (_Float16)acc[mi][ni][r];
        }
      }
    }
}

// ---------------------------------------------------------------------------
// 4) Flash attention. Block = 4 waves = 64 queries; key tiles of 32; hd = 64.
//    K tile (32x64) and V tile (64x32) staged in LDS, shared by all 4 waves
//    (4x less L2 traffic), double buffered with async global->LDS copies.
//    8 WMMAs per 32-key tile per wave. Online softmax in f32 via half-wave
//    shuffles (a C/D-layout row lives in one VGPR across one 16-lane group).
// ---------------------------------------------------------------------------
__global__ void flash_attn_kernel(const _Float16* __restrict__ Q,
                                  const _Float16* __restrict__ Kt,
                                  const _Float16* __restrict__ Vt,
                                  _Float16* __restrict__ AO) {
  __shared__ _Float16 kbuf[2][32 * 64];   // [stage][key_local][c]
  __shared__ _Float16 vbuf[2][64 * 32];   // [stage][c][key_local]
  __shared__ _Float16 pbuf[4][16 * 32];   // per-wave P relayout buffer

  int t = threadIdx.x;
  int wave = t >> 5, lane = t & 31;
  int hi = lane >> 4, ln = lane & 15;
  int b = blockIdx.z, h = blockIdx.y;
  int qb = blockIdx.x * 64 + wave * 16;

  const _Float16* Qh = Q  + ((size_t)b * NH + h) * (size_t)LSP * HD;
  const _Float16* Kh = Kt + ((size_t)b * NH + h) * (size_t)LSP * HD;
  const _Float16* Vh = Vt + ((size_t)b * NH + h) * (size_t)HD * LSP;

  // Query A-fragments (rows qb..qb+15, K split 0..31 / 32..63), reused all loop
  v16h aq0 = load_afrag(Qh + (size_t)(qb + ln) * HD + 0,  hi);
  v16h aq1 = load_afrag(Qh + (size_t)(qb + ln) * HD + 32, hi);

  v8f O[4] = {};
  float mrun[8], lrun[8];
#pragma unroll
  for (int r = 0; r < 8; ++r) { mrun[r] = -__builtin_inff(); lrun[r] = 0.f; }
  const float sscale = 0.0625f;                     // 1/sqrt(C) = 1/16

  // Stage a 32-key tile: K tile = 256 16B chunks, V tile = 256 16B chunks;
  // 128 threads -> 2 + 2 chunks (4 async b128 copies) per thread.
  auto stage_kv = [&](int s, int kb) {
#pragma unroll
    for (int j = 0; j < 2; ++j) {
      int ck = t + j * 128;  // K chunk: key = ck>>3, col = (ck&7)*8
      copy16_g2l(Kh + (size_t)(kb + (ck >> 3)) * HD + (ck & 7) * 8,
                 &kbuf[s][ck * 8]);
      int cv = t + j * 128;  // V chunk: c = cv>>2, col = (cv&3)*8
      copy16_g2l(Vh + (size_t)(cv >> 2) * LSP + kb + (cv & 3) * 8,
                 &vbuf[s][cv * 8]);
    }
  };

  stage_kv(0, 0);                                   // preload first tile

  for (int it = 0, kb = 0; kb < LSP; ++it, kb += 32) {
    int s = it & 1;
    bool more = (kb + 32 < LSP);
    if (more) {
      stage_kv(s ^ 1, kb + 32);                     // issue next tile's copies
      wait_async<4>();                              // current tile done; next in flight
    } else {
      wait_async<0>();
    }
    __syncthreads();                                // staged tile visible block-wide

    // ---- scores S (16q x 32k), contraction over hd=64 -> 4 WMMAs ----
    v8f s0 = {}, s1 = {};
    {
      v16h bk;
      bk = load_bfrag(&kbuf[s][(ln)      * 64 + 0  + 16 * hi]);
      s0 = WMMA_F16(aq0, bk, s0);
      bk = load_bfrag(&kbuf[s][(ln)      * 64 + 32 + 16 * hi]);
      s0 = WMMA_F16(aq1, bk, s0);
      bk = load_bfrag(&kbuf[s][(16 + ln) * 64 + 0  + 16 * hi]);
      s1 = WMMA_F16(aq0, bk, s1);
      bk = load_bfrag(&kbuf[s][(16 + ln) * 64 + 32 + 16 * hi]);
      s1 = WMMA_F16(aq1, bk, s1);
    }

    // ---- online softmax; write P (f16) into LDS in [q][k] order ----
#pragma unroll
    for (int r = 0; r < 8; ++r) {
      float x0 = s0[r] * sscale, x1 = s1[r] * sscale;
      float mx = fmaxf(x0, x1);
      mx = fmaxf(mx, __shfl_xor(mx, 1, 32));
      mx = fmaxf(mx, __shfl_xor(mx, 2, 32));
      mx = fmaxf(mx, __shfl_xor(mx, 4, 32));
      mx = fmaxf(mx, __shfl_xor(mx, 8, 32));        // row max within 16-lane group
      float mnew  = fmaxf(mrun[r], mx);
      float scale = __expf(mrun[r] - mnew);
      float p0 = __expf(x0 - mnew), p1 = __expf(x1 - mnew);
      float rsum = p0 + p1;
      rsum += __shfl_xor(rsum, 1, 32);
      rsum += __shfl_xor(rsum, 2, 32);
      rsum += __shfl_xor(rsum, 4, 32);
      rsum += __shfl_xor(rsum, 8, 32);
      lrun[r] = lrun[r] * scale + rsum;
      mrun[r] = mnew;
      O[0][r] *= scale; O[1][r] *= scale; O[2][r] *= scale; O[3][r] *= scale;
      int row = r + 8 * hi;                         // C/D layout: row = r + 8*hi
      pbuf[wave][row * 32 + ln]      = (_Float16)p0;
      pbuf[wave][row * 32 + 16 + ln] = (_Float16)p1;
    }
    __syncthreads();                                // P store -> A-fragment reload

    // ---- reload P as an A-fragment (16q x 32k) ----
    v16h ap;
    *(h8*)&ap       = *(h8*)&pbuf[wave][ln * 32 + 16 * hi];
    *((h8*)&ap + 1) = *(h8*)&pbuf[wave][ln * 32 + 16 * hi + 8];

    // ---- O += P * V  (4 WMMAs, 16 channels each) ----
#pragma unroll
    for (int f = 0; f < 4; ++f) {
      v16h bv = load_bfrag(&vbuf[s][(f * 16 + ln) * 32 + 16 * hi]);
      O[f] = WMMA_F16(ap, bv, O[f]);
    }
    __syncthreads();  // all waves done reading stage s before it is re-staged
  }

  // ---- finalize: divide by row sum, store attn out as [b][l][c] f16 ----
  _Float16* AOb = AO + (size_t)b * LSP * CHN;
#pragma unroll
  for (int f = 0; f < 4; ++f)
#pragma unroll
    for (int r = 0; r < 8; ++r) {
      int q = qb + r + 8 * hi;
      int c = h * HD + f * 16 + ln;
      AOb[(size_t)q * CHN + c] = (_Float16)(O[f][r] / lrun[r]);
    }
}

// ---------------------------------------------------------------------------
// 5) Output projection: out[b][m][l] = Wo[m][k]*AO[l][k] + bo[m] + x[b][m][l]
// ---------------------------------------------------------------------------
__global__ void oproj_gemm_kernel(const _Float16* __restrict__ W,
                                  const _Float16* __restrict__ AO,
                                  const float* __restrict__ bo,
                                  const float* __restrict__ xres,
                                  float* __restrict__ out) {
  int wave = threadIdx.x >> 5, lane = threadIdx.x & 31;
  int hi = lane >> 4, ln = lane & 15;
  int b  = blockIdx.z;
  int mb = blockIdx.y * 64 + (wave & 1) * 32;
  int nb = blockIdx.x * 64 + (wave >> 1) * 32;
  const _Float16* Ab = AO + (size_t)b * LSP * CHN;

  v8f acc[2][2] = {};
  for (int k0 = 0; k0 < CHN; k0 += 32) {
    v16h a0 = load_afrag(W + (size_t)(mb + ln) * CHN + k0, hi);
    v16h a1 = load_afrag(W + (size_t)(mb + 16 + ln) * CHN + k0, hi);
    v16h b0 = load_bfrag(Ab + (size_t)(nb + ln) * CHN + k0 + 16 * hi);
    v16h b1 = load_bfrag(Ab + (size_t)(nb + 16 + ln) * CHN + k0 + 16 * hi);
    acc[0][0] = WMMA_F16(a0, b0, acc[0][0]);
    acc[0][1] = WMMA_F16(a0, b1, acc[0][1]);
    acc[1][0] = WMMA_F16(a1, b0, acc[1][0]);
    acc[1][1] = WMMA_F16(a1, b1, acc[1][1]);
  }

#pragma unroll
  for (int mi = 0; mi < 2; ++mi)
#pragma unroll
    for (int ni = 0; ni < 2; ++ni) {
      int m0 = mb + mi * 16 + 8 * hi;
      int n  = nb + ni * 16 + ln;
#pragma unroll
      for (int r = 0; r < 8; ++r) {
        int m = m0 + r;
        size_t idx = ((size_t)b * CHN + m) * (size_t)LSP + n;
        out[idx] = acc[mi][ni][r] + bo[m] + xres[idx];
      }
    }
}

// ---------------------------------------------------------------------------
extern "C" void kernel_launch(void* const* d_in, const int* in_sizes, int n_in,
                              void* d_out, int out_size, void* d_ws, size_t ws_size,
                              hipStream_t stream) {
  const float* x  = (const float*)d_in[0];
  const float* gw = (const float*)d_in[1];
  const float* gb = (const float*)d_in[2];
  const float* wq = (const float*)d_in[3];
  const float* wk = (const float*)d_in[4];
  const float* wv = (const float*)d_in[5];
  const float* wo = (const float*)d_in[6];
  const float* bo = (const float*)d_in[7];
  _Float16* ws = (_Float16*)d_ws;                   // ~21.5 MB of f16 used
  float* out = (float*)d_out;

  cvt_weights_kernel<<<1024, 256, 0, stream>>>(wq, wk, wv, wo, ws + OFF_WQ);
  groupnorm_kernel<<<BATCH * GROUPS, 256, 0, stream>>>(x, gw, gb, ws + OFF_XN);

  dim3 gg(LSP / 64, CHN / 64, BATCH);               // (64, 4, 2)
  qkv_gemm_kernel<<<gg, 128, 0, stream>>>(ws + OFF_WQ, ws + OFF_XN, ws + OFF_Q, 0);
  qkv_gemm_kernel<<<gg, 128, 0, stream>>>(ws + OFF_WK, ws + OFF_XN, ws + OFF_K, 0);
  qkv_gemm_kernel<<<gg, 128, 0, stream>>>(ws + OFF_WV, ws + OFF_XN, ws + OFF_V, 1);

  flash_attn_kernel<<<dim3(LSP / 64, NH, BATCH), 128, 0, stream>>>(
      ws + OFF_Q, ws + OFF_K, ws + OFF_V, ws + OFF_AO);

  oproj_gemm_kernel<<<gg, 128, 0, stream>>>(ws + OFF_WO, ws + OFF_AO, bo, x, out);
}
